// CustomMultiHeadAttention_70334384439630
// MI455X (gfx1250) — compile-verified, matrix operand order checked by reading, not run on hardware
//
#include <hip/hip_runtime.h>
#include <hip/hip_bf16.h>
#include <math.h>

typedef __attribute__((ext_vector_type(16))) _Float16 v16h;
typedef __attribute__((ext_vector_type(8)))  _Float16 v8h;
typedef __attribute__((ext_vector_type(8)))  float    v8f;
typedef __attribute__((ext_vector_type(4)))  float    v4f;
typedef __attribute__((ext_vector_type(4)))  int      v4i;

#define WMMA_F32_F16(A, B, C) \
  __builtin_amdgcn_wmma_f32_16x16x32_f16(false, (A), false, (B), (short)0, (C), false, false)

static constexpr int Bn = 2;     // batch
static constexpr int Lc = 2048;  // sequence (text+audio)
static constexpr int Dm = 512;   // model dim
static constexpr int Hc = 8;     // heads
static constexpr int Dh = 64;    // head dim

// ---------------- async global->LDS (gfx1250) with safe fallback ------------
#if defined(__has_builtin)
#  if __has_builtin(__builtin_amdgcn_global_load_async_to_lds_b128)
#    define HAS_ASYNC_LDS 1
#  endif
#endif
#ifndef HAS_ASYNC_LDS
#  define HAS_ASYNC_LDS 0
#endif

__device__ __forceinline__ void copy16_to_lds(_Float16* ldst, const _Float16* gsrc) {
#if HAS_ASYNC_LDS
  // signature: (v4i addrspace(1)* src, v4i addrspace(3)* dst, imm offset, imm cpol)
  __builtin_amdgcn_global_load_async_to_lds_b128(
      (__attribute__((address_space(1))) v4i*)(v4i*)const_cast<_Float16*>(gsrc),
      (__attribute__((address_space(3))) v4i*)(v4i*)ldst,
      0, 0);
#else
  *(v8h*)ldst = *(const v8h*)gsrc;
#endif
}

__device__ __forceinline__ void async_wait_all() {
#if HAS_ASYNC_LDS
#  if __has_builtin(__builtin_amdgcn_s_wait_asynccnt)
  __builtin_amdgcn_s_wait_asynccnt(0);
#  else
  asm volatile("s_wait_asynccnt 0x0" ::: "memory");
#  endif
#endif
}

// ---- WMMA fragment loaders (ISA 7.12.2 16-bit A/B layout) -------------------
// Lane l: row = l&15, hi = l>>4. Element i<8 -> K = 8*hi + i ;
// element i>=8 -> K = 16 + 8*hi + (i-8). Two contiguous 8-element runs.
__device__ __forceinline__ v16h frag_from_f32(const float* __restrict__ p0, int stride) {
  int lane = threadIdx.x & 31;
  const float* p = p0 + (size_t)(lane & 15) * stride + (lane >> 4) * 8;
  v4f a0 = *(const v4f*)(p);
  v4f a1 = *(const v4f*)(p + 4);
  v4f a2 = *(const v4f*)(p + 16);
  v4f a3 = *(const v4f*)(p + 20);
  v16h f;
#pragma unroll
  for (int i = 0; i < 4; ++i) {
    f[i]      = (_Float16)a0[i];
    f[4 + i]  = (_Float16)a1[i];
    f[8 + i]  = (_Float16)a2[i];
    f[12 + i] = (_Float16)a3[i];
  }
  return f;
}

// Works for global f16 or LDS f16 (addrspace inferred after inlining -> ds_load_b128).
__device__ __forceinline__ v16h frag_from_f16(const _Float16* p0, int stride) {
  int lane = threadIdx.x & 31;
  const _Float16* p = p0 + (size_t)(lane & 15) * stride + (lane >> 4) * 8;
  v8h lo = *(const v8h*)(p);
  v8h hi = *(const v8h*)(p + 16);
  v16h f;
#pragma unroll
  for (int i = 0; i < 8; ++i) { f[i] = lo[i]; f[8 + i] = hi[i]; }
  return f;
}

__device__ __forceinline__ v8h cvt8(v4f a, v4f b) {
  v8h r;
#pragma unroll
  for (int i = 0; i < 4; ++i) { r[i] = (_Float16)a[i]; r[4 + i] = (_Float16)b[i]; }
  return r;
}

// ---- mask prep: pad[B,L], coef[B,L] ----------------------------------------
__global__ void prep_kernel(const float* __restrict__ tm, const float* __restrict__ am,
                            float* __restrict__ pad, float* __restrict__ coef) {
  int b = blockIdx.x;
  __shared__ float red[256];
  __shared__ float wts[2];
  float tsum = 0.f, asum = 0.f;
  for (int i = threadIdx.x; i < 1024; i += 256) {
    tsum += tm[b * 1024 + i];
    asum += am[b * 1024 + i];
  }
  red[threadIdx.x] = tsum; __syncthreads();
  for (int s = 128; s > 0; s >>= 1) { if (threadIdx.x < s) red[threadIdx.x] += red[threadIdx.x + s]; __syncthreads(); }
  float tl = red[0]; __syncthreads();
  red[threadIdx.x] = asum; __syncthreads();
  for (int s = 128; s > 0; s >>= 1) { if (threadIdx.x < s) red[threadIdx.x] += red[threadIdx.x + s]; __syncthreads(); }
  float al = red[0]; __syncthreads();
  if (threadIdx.x == 0) {
    float tot = tl + al;
    wts[0] = tot / (2.0f * tl);
    wts[1] = tot / (2.0f * al);
  }
  __syncthreads();
  for (int i = threadIdx.x; i < Lc; i += 256) {
    float m = (i < 1024) ? tm[b * 1024 + i] : am[b * 1024 + i - 1024];
    pad[b * Lc + i]  = m;
    coef[b * Lc + i] = m * wts[i >> 10];
  }
}

// ---- QKV projection: C = X @ W^T + bias, store f16 -------------------------
// Block: 256 threads = 8 waves; all waves share one 64-col W panel via LDS.
// blockIdx.x = m-block (8 row-tiles of 16), blockIdx.y = n-group (64 cols).
// transposed==0: out[b,h,l,dh] (Q,K). transposed==1: out[b,h,dh,l] (V^T).
static constexpr int WSTR = 40;  // padded LDS stride (32 data + 8 pad halves)

__global__ void qkv_proj_kernel(const float* __restrict__ X, const float* __restrict__ W,
                                const float* __restrict__ bias, _Float16* __restrict__ out,
                                int transposed) {
  __shared__ _Float16 Wl[64 * WSTR];
  int tid = threadIdx.x;
  int lane = tid & 31;
  int mt = blockIdx.x * 8 + (tid >> 5);  // row tile of 16 over B*L=4096
  int n0 = blockIdx.y * 64;
  int sn = tid >> 2, sk = (tid & 3) * 8;  // staging: 64 rows x 4 chunks of 8
  const float* Xb = X + (size_t)mt * 16 * Dm;
  v8f acc[4] = {};
  for (int k0 = 0; k0 < Dm; k0 += 32) {
    __syncthreads();  // previous panel fully consumed
    {
      const float* wp = W + (size_t)(n0 + sn) * Dm + k0 + sk;
      *(v8h*)&Wl[sn * WSTR + sk] = cvt8(*(const v4f*)wp, *(const v4f*)(wp + 4));
    }
    __syncthreads();
    v16h a = frag_from_f32(Xb + k0, Dm);
#pragma unroll
    for (int j = 0; j < 4; ++j) {
      v16h bfr = frag_from_f16(&Wl[16 * j * WSTR], WSTR);
      acc[j] = WMMA_F32_F16(a, bfr, acc[j]);
    }
  }
  int hi = lane >> 4, nl = lane & 15;
#pragma unroll
  for (int j = 0; j < 4; ++j) {
    int n = n0 + 16 * j + nl;
    int h = n >> 6, dh = n & 63;
    float bn = bias[n];
#pragma unroll
    for (int v = 0; v < 8; ++v) {
      int r = mt * 16 + v + 8 * hi;          // global row in [0,4096)
      int bb = r >> 11, l = r & (Lc - 1);
      float val = acc[j][v] + bn;
      if (transposed)
        out[((size_t)((bb * Hc + h) * Dh + dh)) * Lc + l] = (_Float16)val;
      else
        out[((size_t)((bb * Hc + h) * Lc + l)) * Dh + dh] = (_Float16)val;
    }
  }
}

// ---- flash attention, transposed-score layout ------------------------------
// Block: 256 threads = 8 waves, one (b,h); each wave owns a 16-query tile.
// K/V^T tiles double-buffered in LDS, filled with async global->LDS DMA.
// S^T = K·Q^T keeps the softmax (key) axis in registers; one xor-16 shuffle
// merges the lane pair. P (x coef) repacks in-register into a f16 B-fragment.
static constexpr int KSTR = 72;  // 64 + 8 pad halves
static constexpr int VSTR = 40;  // 32 + 8 pad halves

__global__ void attn_kernel(const _Float16* __restrict__ Qh, const _Float16* __restrict__ Kh,
                            const _Float16* __restrict__ Vt, const float* __restrict__ pad,
                            const float* __restrict__ coef, _Float16* __restrict__ Ao) {
  __shared__ _Float16 Kl[2][32 * KSTR];
  __shared__ _Float16 Vl[2][64 * VSTR];
  int tid = threadIdx.x;
  int lane = tid & 31;
  int bh = blockIdx.x >> 4;       // 0..15 = b*8+h
  int qg = blockIdx.x & 15;       // 16 query groups of 8 tiles
  int qt = qg * 8 + (tid >> 5);   // this wave's 16-query tile
  int b  = bh >> 3;
  int h  = bh & 7;
  int hi = lane >> 4, nl = lane & 15;

  const _Float16* Qb = Qh + ((size_t)bh * Lc + qt * 16) * Dh;
  const _Float16* Kb = Kh + (size_t)bh * Lc * Dh;
  const _Float16* Vb = Vt + (size_t)bh * Dh * Lc;
  const float* padb  = pad  + b * Lc;
  const float* coefb = coef + b * Lc;

  // cooperative tile staging: one b128 per thread per tile
  int krow = tid >> 3, kcol = (tid & 7) * 8;   // K tile [32 keys][64 dh]
  int vrow = tid >> 2, vcol = (tid & 3) * 8;   // V^T tile [64 dh][32 keys]

  // Q fragments (B operand, n = query lane, K = dh), reused across all keys
  v16h q0f = frag_from_f16(Qb, Dh);
  v16h q1f = frag_from_f16(Qb + 32, Dh);
  bool qmask = (padb[qt * 16 + nl] == 0.0f);

  float m_run = -1e30f, s_run = 0.0f;
  v8f O[4] = {};   // O^T tiles: 4 x (16 dh rows x 16 queries)

  // prologue: stage first tile pair into buffer 0
  copy16_to_lds(&Kl[0][krow * KSTR + kcol], Kb + (size_t)krow * Dh + kcol);
  copy16_to_lds(&Vl[0][vrow * VSTR + vcol], Vb + (size_t)vrow * Lc + vcol);

  int cur = 0;
  for (int k0 = 0; k0 < Lc; k0 += 32) {
    async_wait_all();
    __syncthreads();                      // tiles for this iter visible to all
    if (k0 + 32 < Lc) {                   // prefetch next pair into other buffer
      int nxt = cur ^ 1, kn = k0 + 32;
      copy16_to_lds(&Kl[nxt][krow * KSTR + kcol], Kb + (size_t)(kn + krow) * Dh + kcol);
      copy16_to_lds(&Vl[nxt][vrow * VSTR + vcol], Vb + (size_t)vrow * Lc + kn + vcol);
    }

    v8f st[2];
#pragma unroll
    for (int t = 0; t < 2; ++t) {
      v16h ka0 = frag_from_f16(&Kl[cur][16 * t * KSTR], KSTR);
      v16h ka1 = frag_from_f16(&Kl[cur][16 * t * KSTR + 32], KSTR);
      v8f z = {};
      z = WMMA_F32_F16(ka0, q0f, z);
      st[t] = WMMA_F32_F16(ka1, q1f, z);
    }

    // scale + dual masking (query OR key padded -> -10000, as in reference)
    float s[16];
    float lmax = -1e30f;
#pragma unroll
    for (int t = 0; t < 2; ++t)
#pragma unroll
      for (int j = 0; j < 8; ++j) {
        int key = k0 + 16 * t + j + 8 * hi;
        float v = st[t][j] * 0.125f;   // 1/sqrt(64)
        if (qmask || padb[key] == 0.0f) v = -10000.0f;
        s[t * 8 + j] = v;
        lmax = fmaxf(lmax, v);
      }
    float bmax  = fmaxf(lmax, __shfl_xor(lmax, 16, 32));  // lane-pair combine
    float m_new = fmaxf(m_run, bmax);
    float resc  = __expf(m_run - m_new);

    float p[16];
    float lsum = 0.f;
#pragma unroll
    for (int i = 0; i < 16; ++i) { p[i] = __expf(s[i] - m_new); lsum += p[i]; }
    s_run = s_run * resc + lsum + __shfl_xor(lsum, 16, 32);
    m_run = m_new;

    // Fold post-softmax key reweighting coef[key] into P and repack the two
    // 16-key S^T tiles straight into a f16 B-fragment (K = 32 keys).
    v16h pf;
#pragma unroll
    for (int t = 0; t < 2; ++t)
#pragma unroll
      for (int j = 0; j < 8; ++j) {
        int key = k0 + 16 * t + j + 8 * hi;
        pf[t * 8 + j] = (_Float16)(p[t * 8 + j] * coefb[key]);
      }

#pragma unroll
    for (int d = 0; d < 4; ++d) {
      v16h va = frag_from_f16(&Vl[cur][d * 16 * VSTR], VSTR);  // V^T rows = dh
#pragma unroll
      for (int e = 0; e < 8; ++e) O[d][e] *= resc;             // online rescale
      O[d] = WMMA_F32_F16(va, pf, O[d]);                       // O^T += V^T P
    }
    cur ^= 1;
  }

  float inv = 1.0f / s_run;
  int row = b * Lc + qt * 16 + nl;          // row in [0,4096) of attn output
#pragma unroll
  for (int d = 0; d < 4; ++d)
#pragma unroll
    for (int j = 0; j < 8; ++j) {
      int col = h * Dh + d * 16 + j + 8 * hi;
      Ao[(size_t)row * Dm + col] = (_Float16)(O[d][j] * inv);
    }
}

// ---- output projection: out = Ao @ wo^T + bo (f32 out), shared-W LDS -------
__global__ void out_proj_kernel(const _Float16* __restrict__ X, const float* __restrict__ W,
                                const float* __restrict__ bias, float* __restrict__ out) {
  __shared__ _Float16 Wl[64 * WSTR];
  int tid = threadIdx.x;
  int lane = tid & 31;
  int mt = blockIdx.x * 8 + (tid >> 5);
  int n0 = blockIdx.y * 64;
  int sn = tid >> 2, sk = (tid & 3) * 8;
  const _Float16* Xb = X + (size_t)mt * 16 * Dm;
  v8f acc[4] = {};
  for (int k0 = 0; k0 < Dm; k0 += 32) {
    __syncthreads();
    {
      const float* wp = W + (size_t)(n0 + sn) * Dm + k0 + sk;
      *(v8h*)&Wl[sn * WSTR + sk] = cvt8(*(const v4f*)wp, *(const v4f*)(wp + 4));
    }
    __syncthreads();
    v16h a = frag_from_f16(Xb + k0, Dm);
#pragma unroll
    for (int j = 0; j < 4; ++j) {
      v16h bfr = frag_from_f16(&Wl[16 * j * WSTR], WSTR);
      acc[j] = WMMA_F32_F16(a, bfr, acc[j]);
    }
  }
  int hi = lane >> 4, nl = lane & 15;
#pragma unroll
  for (int j = 0; j < 4; ++j) {
    int n = n0 + 16 * j + nl;
    float bn = bias[n];
#pragma unroll
    for (int v = 0; v < 8; ++v) {
      int r = mt * 16 + v + 8 * hi;
      out[(size_t)r * Dm + n] = acc[j][v] + bn;
    }
  }
}

extern "C" void kernel_launch(void* const* d_in, const int* in_sizes, int n_in,
                              void* d_out, int out_size, void* d_ws, size_t ws_size,
                              hipStream_t stream) {
  const float* q  = (const float*)d_in[0];
  const float* k  = (const float*)d_in[1];
  const float* v  = (const float*)d_in[2];
  const float* tm = (const float*)d_in[3];
  const float* am = (const float*)d_in[4];
  // d_in[5] = n_head (constant 8, folded into the kernels)
  const float* wq = (const float*)d_in[6];
  const float* bq = (const float*)d_in[7];
  const float* wk = (const float*)d_in[8];
  const float* bk = (const float*)d_in[9];
  const float* wv = (const float*)d_in[10];
  const float* bv = (const float*)d_in[11];
  const float* wo = (const float*)d_in[12];
  const float* bo = (const float*)d_in[13];
  float* out = (float*)d_out;

  char* ws = (char*)d_ws;
  _Float16* Qh = (_Float16*)(ws);                  // [B,H,L,Dh] f16, 4 MiB
  _Float16* Kh = (_Float16*)(ws + (4u << 20));     // [B,H,L,Dh] f16, 4 MiB
  _Float16* Vt = (_Float16*)(ws + (8u << 20));     // [B,H,Dh,L] f16, 4 MiB
  _Float16* Ao = (_Float16*)(ws + (12u << 20));    // [B*L, Dm]  f16, 4 MiB
  float* pad   = (float*)(ws + (16u << 20));               // [B,L] f32
  float* coef  = (float*)(ws + (16u << 20) + 16384);       // [B,L] f32

  prep_kernel<<<Bn, 256, 0, stream>>>(tm, am, pad, coef);
  dim3 ggemm(32, 8);
  qkv_proj_kernel<<<ggemm, 256, 0, stream>>>(q, wq, bq, Qh, 0);
  qkv_proj_kernel<<<ggemm, 256, 0, stream>>>(k, wk, bk, Kh, 0);
  qkv_proj_kernel<<<ggemm, 256, 0, stream>>>(v, wv, bv, Vt, 1);
  attn_kernel<<<256, 256, 0, stream>>>(Qh, Kh, Vt, pad, coef, Ao);
  out_proj_kernel<<<ggemm, 256, 0, stream>>>(Ao, wo, bo, out);
}
